// E74FixedDecayCell_32968168964193
// MI455X (gfx1250) — matrix operand order repaired and checked by reading, not compiled, not run on hardware
//
#include <hip/hip_runtime.h>
#include <hip/hip_bf16.h>

#define ALPHA_F 0.9f
#define EPS_F   1e-6f
#define T_DIM 4096
#define B_DIM 16
#define D_DIM 1024
#define N_DIM 64
#define P_DIM 192   // 3*n  (k | v | q)

typedef __attribute__((ext_vector_type(16))) __bf16 v16bf;
typedef __attribute__((ext_vector_type(8)))  float  v8f;

// ---------------------------------------------------------------------------
// Kernel 1: proj[m][c] = sum_d x[m][d] * W[c][d]   (M=65536, K=1024, N=192)
// One wave (32 threads) per 16-row strip; 12 N-tiles accumulated in registers.
// bf16 A/B, f32 accumulate via v_wmma_f32_16x16x32_bf16.
// ---------------------------------------------------------------------------
__global__ __launch_bounds__(32)
void proj_gemm_wmma(const float* __restrict__ x,
                    const float* __restrict__ W,
                    float* __restrict__ proj)
{
    const int lane = threadIdx.x;       // 0..31
    const int Ml   = lane & 15;         // M within tile
    const int half = lane >> 4;         // 0 or 1
    const int m0   = blockIdx.x * 16;   // 4096 blocks cover M=65536

    v8f acc[12];
#pragma unroll
    for (int nt = 0; nt < 12; ++nt) acc[nt] = (v8f){0,0,0,0,0,0,0,0};

    const float* xrow = x + (size_t)(m0 + Ml) * D_DIM;

    for (int k0 = 0; k0 < D_DIM; k0 += 32) {
        // ---- A tile 16x32 bf16: lane L holds M=L%16,
        //      element i -> K = half*16 + i  (16 consecutive K)
        const float4* pa0 = (const float4*)(xrow + k0 + half * 16);
        const float4* pa1 = (const float4*)(xrow + k0 + half * 16 + 8);
        float4 f0 = pa0[0], f1 = pa0[1];
        float4 g0 = pa1[0], g1 = pa1[1];
        v16bf a;
        a[0]=(__bf16)f0.x;  a[1]=(__bf16)f0.y;  a[2]=(__bf16)f0.z;  a[3]=(__bf16)f0.w;
        a[4]=(__bf16)f1.x;  a[5]=(__bf16)f1.y;  a[6]=(__bf16)f1.z;  a[7]=(__bf16)f1.w;
        a[8]=(__bf16)g0.x;  a[9]=(__bf16)g0.y;  a[10]=(__bf16)g0.z; a[11]=(__bf16)g0.w;
        a[12]=(__bf16)g1.x; a[13]=(__bf16)g1.y; a[14]=(__bf16)g1.z; a[15]=(__bf16)g1.w;

#pragma unroll
        for (int nt = 0; nt < 12; ++nt) {
            // ---- B tile 32x16 bf16: lane L holds N=L%16,
            //      element i -> K = half*16 + i  (16 consecutive K)
            const float4* pb =
                (const float4*)(W + (size_t)(nt * 16 + Ml) * D_DIM + k0 + half * 16);
            float4 b0 = pb[0], b1 = pb[1], b2 = pb[2], b3 = pb[3];
            v16bf bm;
            bm[0]=(__bf16)b0.x;  bm[1]=(__bf16)b0.y;  bm[2]=(__bf16)b0.z;  bm[3]=(__bf16)b0.w;
            bm[4]=(__bf16)b1.x;  bm[5]=(__bf16)b1.y;  bm[6]=(__bf16)b1.z;  bm[7]=(__bf16)b1.w;
            bm[8]=(__bf16)b2.x;  bm[9]=(__bf16)b2.y;  bm[10]=(__bf16)b2.z; bm[11]=(__bf16)b2.w;
            bm[12]=(__bf16)b3.x; bm[13]=(__bf16)b3.y; bm[14]=(__bf16)b3.z; bm[15]=(__bf16)b3.w;

            acc[nt] = __builtin_amdgcn_wmma_f32_16x16x32_bf16(
                false, a, false, bm, (short)0, acc[nt], false, false);
        }
    }

    // ---- Store: C/D layout: lane L -> N = L%16, rows M = half*8 + r
#pragma unroll
    for (int nt = 0; nt < 12; ++nt) {
#pragma unroll
        for (int r = 0; r < 8; ++r) {
            int row = m0 + half * 8 + r;
            int col = nt * 16 + Ml;
            proj[(size_t)row * P_DIM + col] = acc[nt][r];
        }
    }
}

// ---------------------------------------------------------------------------
// Kernel 2: delta-rule scan. One block per batch, 256 threads.
// Thread (r = tid/4, h = tid%4) owns S[r][h*16 .. h*16+15] in registers.
// 4-lane groups are lane-aligned in wave32 -> shfl_xor butterfly reductions.
// Double-buffered LDS for k|v|q: one barrier per timestep.
// ---------------------------------------------------------------------------
__global__ __launch_bounds__(256)
void delta_scan(const float* __restrict__ proj,
                const float* __restrict__ S0,
                float* __restrict__ out,       // [T][B][64]
                float* __restrict__ Sfinal)    // [B][64][64]
{
    const int b   = blockIdx.x;
    const int tid = threadIdx.x;
    const int r   = tid >> 2;      // row 0..63
    const int h   = tid & 3;       // column quarter 0..3

    __shared__ float lds[2][P_DIM];

    float S[16];
#pragma unroll
    for (int j = 0; j < 16; ++j)
        S[j] = S0[(size_t)b * (N_DIM * N_DIM) + r * N_DIM + h * 16 + j];

    if (tid < P_DIM)
        lds[0][tid] = proj[(size_t)b * P_DIM + tid];

    int par = 0;
    for (int t = 0; t < T_DIM; ++t) {
        __syncthreads();
        if (t + 1 < T_DIM && tid < P_DIM)
            lds[par ^ 1][tid] = proj[(size_t)((t + 1) * B_DIM + b) * P_DIM + tid];

        const float* lk = lds[par];
        const float* lv = lds[par] + 64;
        const float* lq = lds[par] + 128;

        // retrieved_i = (S · k) * inv_norm ; normsq over this group's 16 cols
        float p_ret = 0.f, p_k2 = 0.f;
#pragma unroll
        for (int j = 0; j < 16; ++j) {
            float kj = lk[h * 16 + j];
            p_ret = __fmaf_rn(S[j], kj, p_ret);
            p_k2  = __fmaf_rn(kj, kj, p_k2);
        }
        p_ret += __shfl_xor(p_ret, 1, 32);
        p_ret += __shfl_xor(p_ret, 2, 32);
        p_k2  += __shfl_xor(p_k2, 1, 32);
        p_k2  += __shfl_xor(p_k2, 2, 32);

        float inv = 1.0f / (__builtin_sqrtf(p_k2) + EPS_F);
        float ds  = (lv[r] - p_ret * inv) * inv;   // delta_i * inv_norm

        // S = ALPHA*S + delta ⊗ k_norm ;  Sq_i = S_new · q
        float p_sq = 0.f;
#pragma unroll
        for (int j = 0; j < 16; ++j) {
            float kj = lk[h * 16 + j];
            float s  = __fmaf_rn(ds, kj, ALPHA_F * S[j]);
            S[j] = s;
            p_sq = __fmaf_rn(s, lq[h * 16 + j], p_sq);
        }
        p_sq += __shfl_xor(p_sq, 1, 32);
        p_sq += __shfl_xor(p_sq, 2, 32);

        if (h == 0) {
            float sig = 1.0f / (1.0f + __expf(-p_sq));
            out[((size_t)t * B_DIM + b) * N_DIM + r] = p_sq * p_sq * sig;
        }
        par ^= 1;
    }

#pragma unroll
    for (int j = 0; j < 16; ++j)
        Sfinal[(size_t)b * (N_DIM * N_DIM) + r * N_DIM + h * 16 + j] = S[j];
}

// ---------------------------------------------------------------------------
extern "C" void kernel_launch(void* const* d_in, const int* in_sizes, int n_in,
                              void* d_out, int out_size, void* d_ws, size_t ws_size,
                              hipStream_t stream)
{
    (void)in_sizes; (void)n_in; (void)out_size; (void)ws_size;

    const float* x  = (const float*)d_in[0];   // (T,B,D) f32
    const float* S0 = (const float*)d_in[1];   // (B,n,n) f32 (zeros)
    const float* W  = (const float*)d_in[2];   // (3n,D)  f32

    float* out  = (float*)d_out;                               // outputs (T,B,n)
    float* Sfin = out + (size_t)T_DIM * B_DIM * N_DIM;         // then S_final (B,n,n)
    float* proj = (float*)d_ws;                                // (T*B, 192) f32, 50.3 MB

    proj_gemm_wmma<<<(T_DIM * B_DIM) / 16, 32, 0, stream>>>(x, W, proj);
    delta_scan<<<B_DIM, 256, 0, stream>>>(proj, S0, out, Sfin);
}
